// FusionRQVAE_v1_47923245089219
// MI455X (gfx1250) — matrix operand order repaired and checked by reading, not compiled, not run on hardware
//
#include <hip/hip_runtime.h>
#include <hip/hip_bf16.h>
#include <stdint.h>

typedef __attribute__((ext_vector_type(16))) __bf16 v16bf;
typedef __attribute__((ext_vector_type(8)))  __bf16 v8bf;
typedef __attribute__((ext_vector_type(8)))  float  v8f;
typedef __attribute__((ext_vector_type(4)))  int    v4i;

#define BATCH   65536
#define EDIM    64
#define NCODE   256
#define BETA_F  0.25f

#if defined(__HIP_DEVICE_COMPILE__) && defined(__has_builtin)
#  if __has_builtin(__builtin_amdgcn_global_load_async_to_lds_b128) && \
      __has_builtin(__builtin_amdgcn_s_wait_asynccnt)
#    define USE_ASYNC_LDS 1
#  endif
#endif
#ifndef USE_ASYNC_LDS
#  define USE_ASYNC_LDS 0
#endif

#if USE_ASYNC_LDS
typedef __attribute__((address_space(1))) v4i* gv4i_p;   // global int4*
typedef __attribute__((address_space(3))) v4i* lv4i_p;   // LDS int4*
#endif

static __device__ __forceinline__ __bf16 f2bf(float f) {
  union { float f; uint32_t u; } v; v.f = f;
  uint32_t r = (v.u + 0x7FFFu + ((v.u >> 16) & 1u)) >> 16;
  uint16_t s = (uint16_t)r;
  return __builtin_bit_cast(__bf16, s);
}

static __device__ __forceinline__ void wait_async0() {
#if USE_ASYNC_LDS
  __builtin_amdgcn_s_wait_asynccnt(0);
#endif
}

static __device__ __forceinline__ void copy16_to_lds(const __bf16* gp, __bf16* lp) {
#if USE_ASYNC_LDS
  __builtin_amdgcn_global_load_async_to_lds_b128((gv4i_p)(void*)gp, (lv4i_p)(void*)lp, 0, 0);
#else
  *(v8bf*)lp = *(const v8bf*)gp;
#endif
}

// A-fragment, contiguous K order: lane&15 = M, (lane>>4)*16 = K base (kb*2)
static __device__ __forceinline__ v16bf a_frag(const __bf16* rowp, int kb) {
  v16bf af;
  const v8bf lo = *(const v8bf*)(rowp + 2 * kb);
  const v8bf hi = *(const v8bf*)(rowp + 2 * kb + 8);
  #pragma unroll
  for (int i = 0; i < 8; ++i) { af[i] = lo[i]; af[i + 8] = hi[i]; }
  return af;
}

// ---------------------------------------------------------------------------
// GEMM 128x128 tile: out = act(A[M,K] @ W[K,N] + bias[N]).
//   8 waves in 2x4: wave (wr=w>>2, wc=w&3) computes rows [64*wr,+64) x cols [32*wc,+32)
//   => 8 WMMAs per wave per 32-wide k-step; B-frags loaded once, reused 4x.
//   INBF=true : A bf16, async-to-LDS double-buffered; INBF=false: f32->bf16 staging.
//   Wt pre-transposed bf16 [N,K]; B-fragments read straight from L2.
// ---------------------------------------------------------------------------
template<bool INBF>
__global__ __launch_bounds__(256)
void gemm128(const void* __restrict__ Ain, const __bf16* __restrict__ Wt,
             const float* __restrict__ bias, float* __restrict__ outf,
             __bf16* __restrict__ outb, int N, int K, int relu)
{
  __shared__ __align__(64) __bf16 As[2][128][32];

  const int t    = threadIdx.x;
  const int lane = t & 31;
  const int w    = t >> 5;
  const int tileM = blockIdx.y << 7;
  const int tileN = blockIdx.x << 7;
  const int wr = w >> 2;           // 0..1 -> 64-row half
  const int wc = w & 3;            // 0..3 -> 32-col strip

  v8f acc[4][2] = {};

  // staging index precompute
  const int br  = t >> 2;          // bf16 path: rows br, br+64
  const int bc  = (t & 3) << 3;    // bf16 path: col*8 (16B)
  const int ar  = t >> 3;          // f32 path: rows ar, +32, +64, +96
  const int ac  = (t & 7) << 2;

  const __bf16* Ab = (const __bf16*)Ain;
  const float*  Af = (const float*)Ain;

  auto stage = [&](int k0, int sel) {
    if constexpr (INBF) {
      copy16_to_lds(&Ab[(size_t)(tileM + br) * K + k0 + bc],      &As[sel][br][bc]);
      copy16_to_lds(&Ab[(size_t)(tileM + br + 64) * K + k0 + bc], &As[sel][br + 64][bc]);
    } else {
      #pragma unroll
      for (int q = 0; q < 4; ++q) {
        const int r = ar + q * 32;
        const float4 x = *(const float4*)&Af[(size_t)(tileM + r) * K + k0 + ac];
        As[sel][r][ac + 0] = f2bf(x.x); As[sel][r][ac + 1] = f2bf(x.y);
        As[sel][r][ac + 2] = f2bf(x.z); As[sel][r][ac + 3] = f2bf(x.w);
      }
    }
  };

  stage(0, 0);
  int buf = 0;

  // fragment geometry (contiguous K order)
  const int kb = (lane >> 4) << 3;   // A: K base 0/16 (2*kb in a_frag)
  const int nl = lane & 15;
  const int kh = (lane >> 4) << 4;   // B: K half 0/16
  const __bf16* WtRow0 = &Wt[(size_t)(tileN + (wc << 5) + nl) * K];
  const __bf16* WtRow1 = &Wt[(size_t)(tileN + (wc << 5) + 16 + nl) * K];
  const int mBaseL = (wr << 6) + (lane & 15);   // local row of A frag for mi=0

  for (int k0 = 0; k0 < K; k0 += 32) {
    wait_async0();
    __syncthreads();
    if (k0 + 32 < K) stage(k0 + 32, buf ^ 1);

    const v16bf b0 = *(const v16bf*)&WtRow0[k0 + kh];
    const v16bf b1 = *(const v16bf*)&WtRow1[k0 + kh];
    #pragma unroll
    for (int mi = 0; mi < 4; ++mi) {
      const v16bf af = a_frag(&As[buf][mBaseL + (mi << 4)][0], kb);
      acc[mi][0] = __builtin_amdgcn_wmma_f32_16x16x32_bf16(false, af, false, b0, (short)0, acc[mi][0], false, false);
      acc[mi][1] = __builtin_amdgcn_wmma_f32_16x16x32_bf16(false, af, false, b1, (short)0, acc[mi][1], false, false);
    }
    buf ^= 1;
  }

  // C layout: VGPR r -> M = r + 8*(lane>>4), N = lane&15
  const int n0  = tileN + (wc << 5) + (lane & 15);
  const float bb0 = bias[n0];
  const float bb1 = bias[n0 + 16];
  #pragma unroll
  for (int mi = 0; mi < 4; ++mi) {
    const int mBase = tileM + (wr << 6) + (mi << 4) + ((lane >> 4) << 3);
    #pragma unroll
    for (int r = 0; r < 8; ++r) {
      float v0 = acc[mi][0][r] + bb0;
      float v1 = acc[mi][1][r] + bb1;
      if (relu) { v0 = fmaxf(v0, 0.f); v1 = fmaxf(v1, 0.f); }
      if (outf) {
        outf[(size_t)(mBase + r) * N + n0]      = v0;
        outf[(size_t)(mBase + r) * N + n0 + 16] = v1;
      }
      if (outb) {
        outb[(size_t)(mBase + r) * N + n0]      = f2bf(v0);
        outb[(size_t)(mBase + r) * N + n0 + 16] = f2bf(v1);
      }
    }
  }
}

// ---------------------------------------------------------------------------
// GEMM 64x64 tile (for N==64 layers). Same structure, 2 WMMAs/wave/k-step.
// ---------------------------------------------------------------------------
template<bool INBF>
__global__ __launch_bounds__(256)
void gemm64(const void* __restrict__ Ain, const __bf16* __restrict__ Wt,
            const float* __restrict__ bias, float* __restrict__ outf,
            __bf16* __restrict__ outb, int N, int K, int relu)
{
  __shared__ __align__(64) __bf16 As[2][64][32];

  const int t    = threadIdx.x;
  const int lane = t & 31;
  const int w    = t >> 5;
  const int tileM = blockIdx.y << 6;
  const int tileN = blockIdx.x << 6;
  const int mSlab = (w & 3) << 4;
  const int nHalf = (w >> 2) << 5;

  v8f c0 = {}; v8f c1 = {};

  const int br  = t >> 2;
  const int bc  = (t & 3) << 3;
  const int ar  = t >> 3;
  const int ac  = (t & 7) << 2;

  const __bf16* Ab = (const __bf16*)Ain;
  const float*  Af = (const float*)Ain;

  auto stage = [&](int k0, int sel) {
    if constexpr (INBF) {
      copy16_to_lds(&Ab[(size_t)(tileM + br) * K + k0 + bc], &As[sel][br][bc]);
    } else {
      const float4 x0 = *(const float4*)&Af[(size_t)(tileM + ar)      * K + k0 + ac];
      const float4 x1 = *(const float4*)&Af[(size_t)(tileM + ar + 32) * K + k0 + ac];
      As[sel][ar][ac + 0] = f2bf(x0.x); As[sel][ar][ac + 1] = f2bf(x0.y);
      As[sel][ar][ac + 2] = f2bf(x0.z); As[sel][ar][ac + 3] = f2bf(x0.w);
      As[sel][ar + 32][ac + 0] = f2bf(x1.x); As[sel][ar + 32][ac + 1] = f2bf(x1.y);
      As[sel][ar + 32][ac + 2] = f2bf(x1.z); As[sel][ar + 32][ac + 3] = f2bf(x1.w);
    }
  };

  stage(0, 0);
  int buf = 0;

  const int mA = mSlab + (lane & 15);
  const int kb = (lane >> 4) << 3;
  const int nl = lane & 15;
  const int kh = (lane >> 4) << 4;
  const __bf16* WtRow0 = &Wt[(size_t)(tileN + nHalf + nl) * K];
  const __bf16* WtRow1 = &Wt[(size_t)(tileN + nHalf + 16 + nl) * K];

  for (int k0 = 0; k0 < K; k0 += 32) {
    wait_async0();
    __syncthreads();
    if (k0 + 32 < K) stage(k0 + 32, buf ^ 1);

    const v16bf af = a_frag(&As[buf][mA][0], kb);
    const v16bf b0 = *(const v16bf*)&WtRow0[k0 + kh];
    const v16bf b1 = *(const v16bf*)&WtRow1[k0 + kh];
    c0 = __builtin_amdgcn_wmma_f32_16x16x32_bf16(false, af, false, b0, (short)0, c0, false, false);
    c1 = __builtin_amdgcn_wmma_f32_16x16x32_bf16(false, af, false, b1, (short)0, c1, false, false);
    buf ^= 1;
  }

  const int mBase = tileM + mSlab + ((lane >> 4) << 3);
  const int n0    = tileN + nHalf + (lane & 15);
  const float bb0 = bias[n0];
  const float bb1 = bias[n0 + 16];
  #pragma unroll
  for (int r = 0; r < 8; ++r) {
    float v0 = c0[r] + bb0;
    float v1 = c1[r] + bb1;
    if (relu) { v0 = fmaxf(v0, 0.f); v1 = fmaxf(v1, 0.f); }
    if (outf) {
      outf[(size_t)(mBase + r) * N + n0]      = v0;
      outf[(size_t)(mBase + r) * N + n0 + 16] = v1;
    }
    if (outb) {
      outb[(size_t)(mBase + r) * N + n0]      = f2bf(v0);
      outb[(size_t)(mBase + r) * N + n0 + 16] = f2bf(v1);
    }
  }
}

// ---------------------------------------------------------------------------
// Weight transpose + convert: W[K,N] f32 -> Wt[N,K] bf16 (tiny, once/launch)
// ---------------------------------------------------------------------------
__global__ __launch_bounds__(256)
void transpose_w_bf16(const float* __restrict__ W, __bf16* __restrict__ Wt,
                      int K, int N)
{
  const int n = blockIdx.x;
  for (int k = threadIdx.x; k < K; k += 256)
    Wt[(size_t)n * K + k] = f2bf(W[(size_t)k * N + n]);
}

// ---------------------------------------------------------------------------
// Codebooks: cb[c] = A[lvl] @ B[mod][lvl]  (c = mod*3 + lvl), fp32 + bf16 + norms
// ---------------------------------------------------------------------------
__global__ __launch_bounds__(256)
void build_codebooks(const float* A0, const float* A1, const float* A2,
                     const float* Bt0, const float* Bt1, const float* Bt2,
                     const float* Bv0, const float* Bv1, const float* Bv2,
                     float* __restrict__ cb, __bf16* __restrict__ cbh,
                     float* __restrict__ cbnorm)
{
  const int c   = blockIdx.x;    // 0..5
  const int lvl = c % 3;
  const int mod = c / 3;
  const float* Ap = (lvl == 0) ? A0 : ((lvl == 1) ? A1 : A2);
  const float* Bp = (mod == 0) ? ((lvl == 0) ? Bt0 : ((lvl == 1) ? Bt1 : Bt2))
                               : ((lvl == 0) ? Bv0 : ((lvl == 1) ? Bv1 : Bv2));
  const int n = threadIdx.x;     // code 0..255
  float a[8];
  #pragma unroll
  for (int r = 0; r < 8; ++r) a[r] = Ap[n * 8 + r];
  float norm = 0.f;
  for (int e = 0; e < EDIM; ++e) {
    float s = 0.f;
    #pragma unroll
    for (int r = 0; r < 8; ++r) s += a[r] * Bp[r * EDIM + e];
    const size_t o = ((size_t)c * NCODE + n) * EDIM + e;
    cb[o]  = s;
    cbh[o] = f2bf(s);
    norm += s * s;
  }
  cbnorm[c * NCODE + n] = norm;
}

// ---------------------------------------------------------------------------
// Fused residual VQ: 16 rows/block, 3 levels; distances via WMMA bf16,
// argmin/gather/loss/straight-through in LDS; emits bf16 x_q for decoder.
// ---------------------------------------------------------------------------
__global__ __launch_bounds__(256)
void rq_fused(const float* __restrict__ z, const float* __restrict__ cb,
              const __bf16* __restrict__ cbh, const float* __restrict__ cbnorm,
              __bf16* __restrict__ xqh, int* __restrict__ idx_out,
              float* __restrict__ loss_acc, int cbBase)
{
  __shared__ __align__(64) float  resf[16][64];
  __shared__ __align__(64) __bf16 resb[16][64];
  __shared__ float xqs[16][64];
  __shared__ float dsh[16][256];
  __shared__ float dminS[16][16];
  __shared__ int   iminS[16][16];
  __shared__ int   idxs[16];
  __shared__ float lred[256];

  const int t    = threadIdx.x;
  const int lane = t & 31;
  const int w    = t >> 5;
  const int row0 = blockIdx.x << 4;
  float lloss = 0.f;

  for (int i = t; i < 1024; i += 256) {
    const int m = i >> 6, e = i & 63;
    resf[m][e] = z[(size_t)(row0 + m) * EDIM + e];
    xqs[m][e]  = 0.f;
  }
  __syncthreads();

  for (int lvl = 0; lvl < 3; ++lvl) {
    const int c = cbBase + lvl;
    for (int i = t; i < 1024; i += 256) {
      const int m = i >> 6, e = i & 63;
      resb[m][e] = f2bf(resf[m][e]);
    }
    __syncthreads();

    // G = res @ cb^T for this wave's 32 codes (two 16x16 tiles), K=64 in 2 steps
    v8f acc0 = {}; v8f acc1 = {};
    const int mA    = lane & 15;
    const int kb    = (lane >> 4) << 3;
    const int nl    = lane & 15;
    const int kh    = (lane >> 4) << 4;
    const int code0 = w << 5;           // codes 32w .. 32w+31
    #pragma unroll
    for (int ks = 0; ks < EDIM; ks += 32) {
      const v16bf af = a_frag(&resb[mA][ks], kb);
      const v16bf bA = *(const v16bf*)&cbh[((size_t)c * NCODE + code0 + nl)      * EDIM + ks + kh];
      const v16bf bB = *(const v16bf*)&cbh[((size_t)c * NCODE + code0 + 16 + nl) * EDIM + ks + kh];
      acc0 = __builtin_amdgcn_wmma_f32_16x16x32_bf16(false, af, false, bA, (short)0, acc0, false, false);
      acc1 = __builtin_amdgcn_wmma_f32_16x16x32_bf16(false, af, false, bB, (short)0, acc1, false, false);
    }
    // d = ||cb||^2 - 2*G  (||res||^2 constant per row; irrelevant for argmin)
    #pragma unroll
    for (int r = 0; r < 8; ++r) {
      const int m  = r + ((lane >> 4) << 3);
      const int cA = code0 + (lane & 15);
      const int cB = cA + 16;
      dsh[m][cA] = cbnorm[c * NCODE + cA] - 2.f * acc0[r];
      dsh[m][cB] = cbnorm[c * NCODE + cB] - 2.f * acc1[r];
    }
    __syncthreads();

    // two-stage argmin (first-min tie-break preserved)
    {
      const int m = t >> 4, seg = t & 15;
      float bv = dsh[m][seg * 16]; int bi = seg * 16;
      #pragma unroll
      for (int k = 1; k < 16; ++k) {
        const float v = dsh[m][seg * 16 + k];
        if (v < bv) { bv = v; bi = seg * 16 + k; }
      }
      dminS[m][seg] = bv; iminS[m][seg] = bi;
    }
    __syncthreads();
    if (t < 16) {
      float bv = dminS[t][0]; int bi = iminS[t][0];
      #pragma unroll
      for (int k = 1; k < 16; ++k) {
        const float v = dminS[t][k];
        if (v < bv) { bv = v; bi = iminS[t][k]; }
      }
      idxs[t] = bi;
      idx_out[(size_t)(row0 + t) * 3 + lvl] = bi;
    }
    __syncthreads();

    // gather (exact fp32), loss += (q - res)^2, x_q += q, res -= q
    for (int i = t; i < 1024; i += 256) {
      const int m = i >> 6, e = i & 63;
      const float q  = cb[((size_t)c * NCODE + idxs[m]) * EDIM + e];
      const float rv = resf[m][e];
      const float d  = q - rv;
      lloss += d * d;
      xqs[m][e] += q;
      resf[m][e] = rv - q;
    }
    __syncthreads();
  }

  for (int i = t; i < 1024; i += 256) {
    const int m = i >> 6, e = i & 63;
    xqh[(size_t)(row0 + m) * EDIM + e] = f2bf(xqs[m][e]);
  }
  lred[t] = lloss;
  __syncthreads();
  for (int s = 128; s > 0; s >>= 1) {
    if (t < s) lred[t] += lred[t + s];
    __syncthreads();
  }
  if (t == 0) atomicAdd(loss_acc, lred[0]);
}

__global__ void zero_losses(float* p) {
  if (threadIdx.x < 2) p[threadIdx.x] = 0.f;
}

__global__ void finalize_losses(const float* __restrict__ acc, float* __restrict__ out2) {
  const float scale = (1.f + BETA_F) / ((float)BATCH * (float)EDIM);
  if (threadIdx.x == 0) { out2[0] = acc[0] * scale; out2[1] = acc[1] * scale; }
}

// ---------------------------------------------------------------------------
extern "C" void kernel_launch(void* const* d_in, const int* in_sizes, int n_in,
                              void* d_out, int out_size, void* d_ws, size_t ws_size,
                              hipStream_t stream) {
  (void)n_in; (void)out_size; (void)ws_size;

  const float *xt = nullptr, *xv = nullptr;
  const float *teW[3], *teB[3], *tdW[3], *tdB[3];
  const float *veW[3], *veB[3], *vdW[3], *vdB[3];
  const float *Ap[3], *Btp[3], *Bvp[3];

  if (in_sizes[0] == 2048) {
    // JAX sorted-key flatten: A, B_text, B_vis, text_dec, text_enc, vis_dec, vis_enc, x_text, x_vis
    int i = 0;
    for (int l = 0; l < 3; ++l) Ap[l]  = (const float*)d_in[i++];
    for (int l = 0; l < 3; ++l) Btp[l] = (const float*)d_in[i++];
    for (int l = 0; l < 3; ++l) Bvp[l] = (const float*)d_in[i++];
    for (int l = 0; l < 3; ++l) { tdW[l] = (const float*)d_in[i++]; tdB[l] = (const float*)d_in[i++]; }
    for (int l = 0; l < 3; ++l) { teW[l] = (const float*)d_in[i++]; teB[l] = (const float*)d_in[i++]; }
    for (int l = 0; l < 3; ++l) { vdW[l] = (const float*)d_in[i++]; vdB[l] = (const float*)d_in[i++]; }
    for (int l = 0; l < 3; ++l) { veW[l] = (const float*)d_in[i++]; veB[l] = (const float*)d_in[i++]; }
    xt = (const float*)d_in[i++];
    xv = (const float*)d_in[i++];
  } else {
    // insertion-order flatten: x_text, x_vis, text_enc, text_dec, vis_enc, vis_dec, A, B_text, B_vis
    xt = (const float*)d_in[0];
    xv = (const float*)d_in[1];
    int i = 2;
    for (int l = 0; l < 3; ++l) { teW[l] = (const float*)d_in[i++]; teB[l] = (const float*)d_in[i++]; }
    for (int l = 0; l < 3; ++l) { tdW[l] = (const float*)d_in[i++]; tdB[l] = (const float*)d_in[i++]; }
    for (int l = 0; l < 3; ++l) { veW[l] = (const float*)d_in[i++]; veB[l] = (const float*)d_in[i++]; }
    for (int l = 0; l < 3; ++l) { vdW[l] = (const float*)d_in[i++]; vdB[l] = (const float*)d_in[i++]; }
    for (int l = 0; l < 3; ++l) Ap[l]  = (const float*)d_in[i++];
    for (int l = 0; l < 3; ++l) Btp[l] = (const float*)d_in[i++];
    for (int l = 0; l < 3; ++l) Bvp[l] = (const float*)d_in[i++];
  }

  // Layer dims: {K, N} for each of the 12 GEMMs (enc/dec per modality)
  const int teK[3] = {768, 512, 256}, teN[3] = {512, 256, 64};
  const int tdK[3] = {64, 256, 512},  tdN[3] = {256, 512, 768};
  const int veK[3] = {512, 512, 256}, veN[3] = {512, 256, 64};
  const int vdK[3] = {64, 256, 512},  vdN[3] = {256, 512, 512};

  // Workspace layout (all regions 256B aligned)
  char* ws = (char*)d_ws;
  size_t off = 0;
  auto alloc = [&](size_t bytes) { char* p = ws + off; off += (bytes + 255) & ~(size_t)255; return p; };

  __bf16* h1b = (__bf16*)alloc((size_t)BATCH * 512 * 2);
  __bf16* h2b = (__bf16*)alloc((size_t)BATCH * 256 * 2);
  float*  zb  = (float*)alloc((size_t)BATCH * EDIM * 4);
  __bf16* xqh = (__bf16*)alloc((size_t)BATCH * EDIM * 2);
  float*  cb     = (float*)alloc((size_t)6 * NCODE * EDIM * 4);
  __bf16* cbh    = (__bf16*)alloc((size_t)6 * NCODE * EDIM * 2);
  float*  cbnorm = (float*)alloc((size_t)6 * NCODE * 4);
  float*  lossacc = (float*)alloc(256);

  __bf16 *teWt[3], *tdWt[3], *veWt[3], *vdWt[3];
  for (int l = 0; l < 3; ++l) teWt[l] = (__bf16*)alloc((size_t)teK[l] * teN[l] * 2);
  for (int l = 0; l < 3; ++l) tdWt[l] = (__bf16*)alloc((size_t)tdK[l] * tdN[l] * 2);
  for (int l = 0; l < 3; ++l) veWt[l] = (__bf16*)alloc((size_t)veK[l] * veN[l] * 2);
  for (int l = 0; l < 3; ++l) vdWt[l] = (__bf16*)alloc((size_t)vdK[l] * vdN[l] * 2);

  // Output layout: out_text | out_vis | loss_t | loss_v | idx_t | idx_v
  float* outp     = (float*)d_out;
  float* out_text = outp;
  float* out_vis  = outp + (size_t)BATCH * 768;
  float* loss_out = outp + (size_t)BATCH * (768 + 512);
  int*   idx_t    = (int*)(loss_out + 2);
  int*   idx_v    = idx_t + (size_t)BATCH * 3;

  zero_losses<<<1, 32, 0, stream>>>(lossacc);
  build_codebooks<<<6, 256, 0, stream>>>(Ap[0], Ap[1], Ap[2],
                                         Btp[0], Btp[1], Btp[2],
                                         Bvp[0], Bvp[1], Bvp[2],
                                         cb, cbh, cbnorm);
  for (int l = 0; l < 3; ++l) {
    transpose_w_bf16<<<teN[l], 256, 0, stream>>>(teW[l], teWt[l], teK[l], teN[l]);
    transpose_w_bf16<<<tdN[l], 256, 0, stream>>>(tdW[l], tdWt[l], tdK[l], tdN[l]);
    transpose_w_bf16<<<veN[l], 256, 0, stream>>>(veW[l], veWt[l], veK[l], veN[l]);
    transpose_w_bf16<<<vdN[l], 256, 0, stream>>>(vdW[l], vdWt[l], vdK[l], vdN[l]);
  }

  // Dispatch: 128x128 tile when N % 128 == 0, else 64x64 tile.
  auto Gf32 = [&](const float* Ain, const __bf16* Wtp, const float* bp,
                  float* of, __bf16* ob, int N, int K, int relu) {
    if (N % 128 == 0) {
      dim3 grid(N / 128, BATCH / 128);
      gemm128<false><<<grid, 256, 0, stream>>>((const void*)Ain, Wtp, bp, of, ob, N, K, relu);
    } else {
      dim3 grid(N / 64, BATCH / 64);
      gemm64<false><<<grid, 256, 0, stream>>>((const void*)Ain, Wtp, bp, of, ob, N, K, relu);
    }
  };
  auto Gbf = [&](const __bf16* Ain, const __bf16* Wtp, const float* bp,
                 float* of, __bf16* ob, int N, int K, int relu) {
    if (N % 128 == 0) {
      dim3 grid(N / 128, BATCH / 128);
      gemm128<true><<<grid, 256, 0, stream>>>((const void*)Ain, Wtp, bp, of, ob, N, K, relu);
    } else {
      dim3 grid(N / 64, BATCH / 64);
      gemm64<true><<<grid, 256, 0, stream>>>((const void*)Ain, Wtp, bp, of, ob, N, K, relu);
    }
  };

  // ---- text path ----
  Gf32(xt,  teWt[0], teB[0], nullptr, h1b, 512, 768, 1);
  Gbf(h1b,  teWt[1], teB[1], nullptr, h2b, 256, 512, 1);
  Gbf(h2b,  teWt[2], teB[2], zb, nullptr, 64, 256, 0);
  rq_fused<<<BATCH / 16, 256, 0, stream>>>(zb, cb, cbh, cbnorm, xqh, idx_t, lossacc + 0, 0);
  Gbf(xqh,  tdWt[0], tdB[0], nullptr, h2b, 256, 64, 1);
  Gbf(h2b,  tdWt[1], tdB[1], nullptr, h1b, 512, 256, 1);
  Gbf(h1b,  tdWt[2], tdB[2], out_text, nullptr, 768, 512, 0);

  // ---- vis path ----
  Gf32(xv,  veWt[0], veB[0], nullptr, h1b, 512, 512, 1);
  Gbf(h1b,  veWt[1], veB[1], nullptr, h2b, 256, 512, 1);
  Gbf(h2b,  veWt[2], veB[2], zb, nullptr, 64, 256, 0);
  rq_fused<<<BATCH / 16, 256, 0, stream>>>(zb, cb, cbh, cbnorm, xqh, idx_v, lossacc + 1, 3);
  Gbf(xqh,  vdWt[0], vdB[0], nullptr, h2b, 256, 64, 1);
  Gbf(h2b,  vdWt[1], vdB[1], nullptr, h1b, 512, 256, 1);
  Gbf(h1b,  vdWt[2], vdB[2], out_vis, nullptr, 512, 512, 0);

  finalize_losses<<<1, 1, 0, stream>>>(lossacc, loss_out);
}
